// GraphFormer_32822140076792
// MI455X (gfx1250) — matrix operand order repaired and checked by reading, not compiled, not run on hardware
//
#include <hip/hip_runtime.h>

#define NN 50000
#define EE 800000
#define HEADS 4
#define CDIM 32
#define HC 128          // HEADS*CDIM
#define RT_TILES 3125   // NN/16

typedef __attribute__((ext_vector_type(16))) __bf16 v16bf;
typedef __attribute__((ext_vector_type(8)))  float  v8f;

// ---------------- zero fill (graph-capture-safe re-init of accumulators) ---
__global__ void gf_zero(float* __restrict__ p, long n) {
    long i = blockIdx.x * (long)blockDim.x + threadIdx.x;
    if (i < n) p[i] = 0.0f;
}

// ---------------- bf16 WMMA GEMM: C[NN x P] = A[NN x K] @ W[K x P] + bias --
// one wave computes a 16 x (16*NT) output tile; K,P compile-time; NN % 16 == 0.
template<int K, int P, int NT>
__global__ __launch_bounds__(32)
void gf_gemm(const float* __restrict__ A, const float* __restrict__ W,
             const float* __restrict__ bias, float* __restrict__ C) {
    const int tm   = blockIdx.x << 4;
    const int tn0  = blockIdx.y * (16 * NT);
    const int lane = threadIdx.x;
    const int l15  = lane & 15;
    const int hi   = lane >> 4;          // 0 or 1

    v8f acc[NT];
    #pragma unroll
    for (int t = 0; t < NT; ++t) acc[t] = (v8f){};

    const int rowA = tm + l15;
    #pragma unroll
    for (int kk = 0; kk < K; kk += 32) {
        // ---- A operand (shared by all NT column tiles) -------------------
        // element i -> K index kk + 16*hi + i: one 16-float run.
        const float* Ap = A + (long)rowA * K + kk + (hi << 4);
        float4 a0 = *(const float4*)(Ap + 0);
        float4 a1 = *(const float4*)(Ap + 4);
        float4 a2 = *(const float4*)(Ap + 8);
        float4 a3 = *(const float4*)(Ap + 12);
        v16bf a;
        a[0]  = (__bf16)a0.x; a[1]  = (__bf16)a0.y; a[2]  = (__bf16)a0.z; a[3]  = (__bf16)a0.w;
        a[4]  = (__bf16)a1.x; a[5]  = (__bf16)a1.y; a[6]  = (__bf16)a1.z; a[7]  = (__bf16)a1.w;
        a[8]  = (__bf16)a2.x; a[9]  = (__bf16)a2.y; a[10] = (__bf16)a2.z; a[11] = (__bf16)a2.w;
        a[12] = (__bf16)a3.x; a[13] = (__bf16)a3.y; a[14] = (__bf16)a3.z; a[15] = (__bf16)a3.w;

        // ---- NT B operands + WMMAs --------------------------------------
        #pragma unroll
        for (int t = 0; t < NT; ++t) {
            const float* Wp = W + (long)(kk + (hi << 4)) * P + (tn0 + t * 16 + l15);
            v16bf b;
            #pragma unroll
            for (int i = 0; i < 16; ++i)
                b[i] = (__bf16)Wp[i * P];
            acc[t] = __builtin_amdgcn_wmma_f32_16x16x32_bf16(
                         false, a, false, b, (short)0, acc[t], false, false);
        }
    }

    #pragma unroll
    for (int t = 0; t < NT; ++t) {
        const int col = tn0 + t * 16 + l15;
        const float bv = bias[col];
        #pragma unroll
        for (int r = 0; r < 8; ++r) {
            int row = tm + r + (hi << 3);
            C[(long)row * P + col] = acc[t][r] + bv;
        }
    }
}

// ---------------- monotone float <-> uint encoding for atomicMax -----------
__device__ __forceinline__ unsigned gf_enc(float f) {
    unsigned b = __float_as_uint(f);
    return (b & 0x80000000u) ? ~b : (b | 0x80000000u);
}
__device__ __forceinline__ float gf_dec(unsigned u) {
    unsigned b = (u & 0x80000000u) ? (u ^ 0x80000000u) : ~u;
    return __uint_as_float(b);
}

// ---------------- per-edge attention logit + segment max -------------------
__global__ void gf_edge_alpha(const int* __restrict__ src, const int* __restrict__ dst,
                              const float* __restrict__ q, const float* __restrict__ k,
                              float* __restrict__ alpha, unsigned* __restrict__ m) {
    long idx = blockIdx.x * (long)blockDim.x + threadIdx.x;
    if (idx >= (long)EE * HEADS) return;
    int  h = idx & (HEADS - 1);
    long e = idx >> 2;
    int  j = src[e], i = dst[e];
    const float4* qp = (const float4*)(q + (long)i * HC + h * CDIM);
    const float4* kp = (const float4*)(k + (long)j * HC + h * CDIM);
    float acc = 0.0f;
    #pragma unroll
    for (int c = 0; c < CDIM / 4; ++c) {
        float4 qv = qp[c], kv = kp[c];
        acc += qv.x * kv.x + qv.y * kv.y + qv.z * kv.z + qv.w * kv.w;
    }
    acc *= 0.17677669529663687f;               // 1/sqrt(32)
    alpha[idx] = acc;
    atomicMax(m + (long)i * HEADS + h, gf_enc(acc));
}

// ---------------- fused exp + unnormalized scatter-accumulate --------------
__global__ void gf_edge_accum(const int* __restrict__ src, const int* __restrict__ dst,
                              const float* __restrict__ v, const float* __restrict__ alpha,
                              const unsigned* __restrict__ m, float* __restrict__ s,
                              float* __restrict__ agg) {
    long idx = blockIdx.x * (long)blockDim.x + threadIdx.x;
    if (idx >= (long)EE * HC) return;
    int  c = idx & (CDIM - 1);
    int  h = (idx >> 5) & (HEADS - 1);
    long e = idx >> 7;
    int  j = src[e], i = dst[e];
    float mx = gf_dec(m[(long)i * HEADS + h]);
    float a  = __expf(alpha[e * HEADS + h] - mx);
    if (c == 0) atomicAdd(s + (long)i * HEADS + h, a);
    atomicAdd(agg + (long)i * HC + h * CDIM + c,
              a * v[(long)j * HC + h * CDIM + c]);
}

// ---------------- normalize, mean over heads, + skip (+ReLU) ---------------
__global__ void gf_node_finish(const float* __restrict__ agg, const float* __restrict__ s,
                               const float* __restrict__ skip, float* __restrict__ out,
                               int do_relu) {
    long idx = blockIdx.x * (long)blockDim.x + threadIdx.x;
    if (idx >= (long)NN * CDIM) return;
    int  c = idx & (CDIM - 1);
    long n = idx >> 5;
    float acc = 0.0f;
    #pragma unroll
    for (int h = 0; h < HEADS; ++h) {
        float sv = s[n * HEADS + h];
        if (sv > 0.0f) acc += agg[n * HC + h * CDIM + c] / sv;
    }
    float o = acc * 0.25f + skip[idx];
    if (do_relu) o = fmaxf(o, 0.0f);
    out[idx] = o;
}

// ---------------------------------------------------------------------------
extern "C" void kernel_launch(void* const* d_in, const int* in_sizes, int n_in,
                              void* d_out, int out_size, void* d_ws, size_t ws_size,
                              hipStream_t stream) {
    const float* x   = (const float*)d_in[0];
    const int*   ei  = (const int*)d_in[1];
    const int*   src = ei;
    const int*   dst = ei + EE;
    const float* Wq1 = (const float*)d_in[2];  const float* bq1 = (const float*)d_in[3];
    const float* Wk1 = (const float*)d_in[4];  const float* bk1 = (const float*)d_in[5];
    const float* Wv1 = (const float*)d_in[6];  const float* bv1 = (const float*)d_in[7];
    const float* Ws1 = (const float*)d_in[8];  const float* bs1 = (const float*)d_in[9];
    const float* Wq2 = (const float*)d_in[10]; const float* bq2 = (const float*)d_in[11];
    const float* Wk2 = (const float*)d_in[12]; const float* bk2 = (const float*)d_in[13];
    const float* Wv2 = (const float*)d_in[14]; const float* bv2 = (const float*)d_in[15];
    const float* Ws2 = (const float*)d_in[16]; const float* bs2 = (const float*)d_in[17];
    float* out = (float*)d_out;

    // workspace carve-up (floats). agg/m/s contiguous so one zero-fill covers them.
    float*    ws    = (float*)d_ws;
    float*    q     = ws;                       // NN*HC
    float*    k     = q    + (long)NN * HC;     // NN*HC
    float*    v     = k    + (long)NN * HC;     // NN*HC
    float*    agg   = v    + (long)NN * HC;     // NN*HC
    unsigned* m     = (unsigned*)(agg + (long)NN * HC);   // NN*HEADS
    float*    s     = (float*)m + (long)NN * HEADS;       // NN*HEADS
    float*    alpha = s    + (long)NN * HEADS;  // EE*HEADS
    float*    skip  = alpha + (long)EE * HEADS; // NN*CDIM
    float*    h1    = skip + (long)NN * CDIM;   // NN*CDIM

    const long nzero = (long)NN * (HC + 2 * HEADS);       // agg + m + s
    const dim3 blk(256);
    const dim3 gz((unsigned)((nzero + 255) / 256));
    const dim3 ga((unsigned)(((long)EE * HEADS + 255) / 256));
    const dim3 gc((unsigned)(((long)EE * HC + 255) / 256));
    const dim3 gn((unsigned)(((long)NN * CDIM + 255) / 256));

    // ---------------- layer 1 (K=64) ----------------
    gf_gemm<64, 128, 4><<<dim3(RT_TILES, 2), 32, 0, stream>>>(x, Wq1, bq1, q);
    gf_gemm<64, 128, 4><<<dim3(RT_TILES, 2), 32, 0, stream>>>(x, Wk1, bk1, k);
    gf_gemm<64, 128, 4><<<dim3(RT_TILES, 2), 32, 0, stream>>>(x, Wv1, bv1, v);
    gf_gemm<64, 32, 2><<<dim3(RT_TILES, 1), 32, 0, stream>>>(x, Ws1, bs1, skip);
    gf_zero<<<gz, blk, 0, stream>>>(agg, nzero);
    gf_edge_alpha<<<ga, blk, 0, stream>>>(src, dst, q, k, alpha, m);
    gf_edge_accum<<<gc, blk, 0, stream>>>(src, dst, v, alpha, m, s, agg);
    gf_node_finish<<<gn, blk, 0, stream>>>(agg, s, skip, h1, 1);

    // ---------------- layer 2 (K=32) ----------------
    gf_gemm<32, 128, 4><<<dim3(RT_TILES, 2), 32, 0, stream>>>(h1, Wq2, bq2, q);
    gf_gemm<32, 128, 4><<<dim3(RT_TILES, 2), 32, 0, stream>>>(h1, Wk2, bk2, k);
    gf_gemm<32, 128, 4><<<dim3(RT_TILES, 2), 32, 0, stream>>>(h1, Wv2, bv2, v);
    gf_gemm<32, 32, 2><<<dim3(RT_TILES, 1), 32, 0, stream>>>(h1, Ws2, bs2, skip);
    gf_zero<<<gz, blk, 0, stream>>>(agg, nzero);
    gf_edge_alpha<<<ga, blk, 0, stream>>>(src, dst, q, k, alpha, m);
    gf_edge_accum<<<gc, blk, 0, stream>>>(src, dst, v, alpha, m, s, agg);
    gf_node_finish<<<gn, blk, 0, stream>>>(agg, s, skip, out, 0);
}
